// BinaryTreeCRF_50388556316896
// MI455X (gfx1250) — compile-verified, matrix operand order checked by reading, not run on hardware
//
#include <hip/hip_runtime.h>

#define B_DIM 16
#define N_DIM 8191
#define D_DIM 512
#define L_DIM 32
#define NLEAF 4096
#define ROWS_TOTAL (B_DIM * N_DIM) /* 131056, exactly 8191 tiles of 16 */

typedef __attribute__((ext_vector_type(16))) __bf16          bf16x16;
typedef __attribute__((ext_vector_type(16))) unsigned short  u16x16;
typedef __attribute__((ext_vector_type(8)))  unsigned int    u32x8;
typedef __attribute__((ext_vector_type(8)))  float           f32x8;

// One-time (init kernels): round-to-nearest-even f32 -> bf16
static __device__ __forceinline__ unsigned short f2bf(float f) {
  unsigned u = __builtin_bit_cast(unsigned, f);
  u += 0x7FFFu + ((u >> 16) & 1u);
  return (unsigned short)(u >> 16);
}

// Hot path: pack two f32 into a bf16x2 dword with ONE v_perm_b32 (truncation).
// Result bytes: [hi.b3, hi.b2, lo.b3, lo.b2]  -> element0 = lo, element1 = hi.
static __device__ __forceinline__ unsigned packbf2(float lo, float hi) {
  return __builtin_amdgcn_perm(__builtin_bit_cast(unsigned, hi),
                               __builtin_bit_cast(unsigned, lo),
                               0x07060302u);
}

static __device__ __forceinline__ bf16x16 load_frag16(const unsigned short* p) {
  const uint4* q = reinterpret_cast<const uint4*>(p);
  uint4 a = q[0];
  uint4 b = q[1];
  u32x8 v = {a.x, a.y, a.z, a.w, b.x, b.y, b.z, b.w};
  return __builtin_bit_cast(bf16x16, v);
}

// K-slot (within a 32-wide K block) held by element e of a lane with hi = lane>>4
static __device__ __forceinline__ int klocal(int e, int hi) {
  return (e < 8 ? e : e + 8) + hi * 8;
}

// ---------------------------------------------------------------------------
// Init kernel 1: swizzle W_pred (512x32 f32) into bf16 WMMA B-fragment layout:
// flat index ((ct*16 + ks)*32 + lane)*16 + e ; 2 ctiles x 16 ksteps.
// ---------------------------------------------------------------------------
__global__ void wswz_kernel(const float* __restrict__ W, unsigned short* __restrict__ wswz) {
  int tid = blockIdx.x * 128 + threadIdx.x;  // 0..1023
  int lane = tid & 31;
  int ks   = (tid >> 5) & 15;
  int ct   = tid >> 9;
  int m16 = lane & 15, hi = lane >> 4;
  int c = ct * 16 + m16;
  #pragma unroll
  for (int e = 0; e < 16; ++e) {
    int k = ks * 32 + klocal(e, hi);
    wswz[(size_t)tid * 16 + e] = f2bf(W[(size_t)k * L_DIM + c]);
  }
}

// ---------------------------------------------------------------------------
// Init kernel 2: tmax = max(trans); etswz[p][lr] = bf16(exp(trans[p,l,r]-tmax))
// stored transposed as B-fragments: flat ((ct*32 + ks)*32 + lane)*16 + e,
// K = l*32+r (1024), N = p (32).
// ---------------------------------------------------------------------------
__global__ __launch_bounds__(1024) void et_kernel(const float* __restrict__ trans,
                                                  unsigned short* __restrict__ etswz,
                                                  float* __restrict__ tmaxp) {
  __shared__ float red[1024];
  int tid = threadIdx.x;
  float mv = -3.402823466e38f;
  for (int t = tid; t < L_DIM * L_DIM * L_DIM; t += 1024) mv = fmaxf(mv, trans[t]);
  red[tid] = mv;
  __syncthreads();
  for (int s = 512; s > 0; s >>= 1) {
    if (tid < s) red[tid] = fmaxf(red[tid], red[tid + s]);
    __syncthreads();
  }
  float tmax = red[0];
  if (tid == 0) tmaxp[0] = tmax;
  #pragma unroll
  for (int i = 0; i < 32; ++i) {
    int s = tid * 32 + i;                    // 32768 bf16 total
    int e    = s & 15;
    int lane = (s >> 4) & 31;
    int ks   = (s >> 9) & 31;
    int ct   = s >> 14;
    int p = ct * 16 + (lane & 15);
    int k = ks * 32 + klocal(e, lane >> 4);  // k = l*32 + r
    etswz[s] = f2bf(__expf(trans[(size_t)p * 1024 + k] - tmax));
  }
}

// ---------------------------------------------------------------------------
// Phase 1: emission = h @ W + b.  One wave per 16-row tile, N=32 (2 ctiles),
// K=512 as 16 bf16 WMMA steps.  Bandwidth-bound on h (268 MB).
// ---------------------------------------------------------------------------
__global__ __launch_bounds__(128) void emission_kernel(
    const float* __restrict__ h, const unsigned short* __restrict__ wswz,
    const float* __restrict__ bpred, float* __restrict__ em)
{
  int wave = threadIdx.x >> 5, lane = threadIdx.x & 31;
  int tile = blockIdx.x * 4 + wave;
  if (tile >= ROWS_TOTAL / 16) return;       // wave-uniform guard
  int m16 = lane & 15, hi = lane >> 4;
  size_t row = (size_t)tile * 16 + m16;
  const float* hp = h + row * D_DIM + hi * 8;
  f32x8 c0 = {}, c1 = {};
  #pragma unroll
  for (int ks = 0; ks < 16; ++ks) {
    const float4* q = (const float4*)(hp + ks * 32);
    float4 x0 = q[0], x1 = q[1], y0 = q[4], y1 = q[5];
    u32x8 ap;
    ap[0] = packbf2(x0.x, x0.y);
    ap[1] = packbf2(x0.z, x0.w);
    ap[2] = packbf2(x1.x, x1.y);
    ap[3] = packbf2(x1.z, x1.w);
    ap[4] = packbf2(y0.x, y0.y);
    ap[5] = packbf2(y0.z, y0.w);
    ap[6] = packbf2(y1.x, y1.y);
    ap[7] = packbf2(y1.z, y1.w);
    bf16x16 a  = __builtin_bit_cast(bf16x16, ap);
    bf16x16 b0 = load_frag16(wswz + ((size_t)ks * 32 + lane) * 16);
    bf16x16 b1 = load_frag16(wswz + ((size_t)(16 + ks) * 32 + lane) * 16);
    c0 = __builtin_amdgcn_wmma_f32_16x16x32_bf16(false, a, false, b0, (short)0, c0, false, false);
    c1 = __builtin_amdgcn_wmma_f32_16x16x32_bf16(false, a, false, b1, (short)0, c1, false, false);
  }
  float b0v = bpred[m16], b1v = bpred[16 + m16];
  #pragma unroll
  for (int j = 0; j < 8; ++j) {
    size_t orow = (size_t)tile * 16 + j + hi * 8;   // C/D layout: VGPR j -> M=j / j+8
    em[orow * L_DIM + m16]      = c0[j] + b0v;
    em[orow * L_DIM + 16 + m16] = c1[j] + b1v;
  }
}

// ---------------------------------------------------------------------------
// Phase 2 level body: one wave handles 16 nodes.
//   comb[p] = tmax + lmax + rmax + log( sum_{l,r} ET[p,l,r] el[l] er[r] )
// done as a K=1024 bf16 WMMA GEMM: A[row, l*32+r] = el[l]*er[r], B = ET^T.
// ---------------------------------------------------------------------------
static __device__ __forceinline__ void level_body(
    bool active, int tile, int lane,
    const float* __restrict__ src, int srcB, int srcOff,
    const float* __restrict__ emission,
    const unsigned short* __restrict__ etswz,
    float tmax, float* __restrict__ dst, int M, int parentStart,
    float* smaxRow)
{
  int m16 = lane & 15, hi = lane >> 4;
  float el[32];
  float er[16];
  if (active) {
    int row = tile * 16 + m16;
    int b = row / M;
    int m = row - b * M;
    const float* Lp = src + (size_t)b * srcB + (size_t)(srcOff + 2 * m) * L_DIM;
    const float4* L4 = (const float4*)Lp;
    const float4* R4 = (const float4*)(Lp + L_DIM);
    float lv[32], rv[32];
    #pragma unroll
    for (int q = 0; q < 8; ++q) {
      float4 t = L4[q];
      lv[4*q+0] = t.x; lv[4*q+1] = t.y; lv[4*q+2] = t.z; lv[4*q+3] = t.w;
      float4 u = R4[q];
      rv[4*q+0] = u.x; rv[4*q+1] = u.y; rv[4*q+2] = u.z; rv[4*q+3] = u.w;
    }
    float lmax = lv[0], rmax = rv[0];
    #pragma unroll
    for (int k = 1; k < 32; ++k) { lmax = fmaxf(lmax, lv[k]); rmax = fmaxf(rmax, rv[k]); }
    #pragma unroll
    for (int k = 0; k < 32; ++k) el[k] = __expf(lv[k] - lmax);
    #pragma unroll
    for (int e = 0; e < 16; ++e) {
      int k0 = (e < 8) ? e : e + 8;
      float v = hi ? rv[k0 + 8] : rv[k0];    // r-slot for this lane half
      er[e] = __expf(v - rmax);
    }
    if (hi == 0) smaxRow[m16] = lmax + rmax;
  }
  __syncthreads();
  if (active) {
    f32x8 c0 = {}, c1 = {};
    #pragma unroll
    for (int ks = 0; ks < 32; ++ks) {        // l = ks; per step r spans 0..31
      float elk = el[ks];
      u32x8 ap;
      #pragma unroll
      for (int e2 = 0; e2 < 8; ++e2)
        ap[e2] = packbf2(elk * er[2 * e2], elk * er[2 * e2 + 1]);
      bf16x16 a  = __builtin_bit_cast(bf16x16, ap);
      bf16x16 b0 = load_frag16(etswz + ((size_t)ks * 32 + lane) * 16);
      bf16x16 b1 = load_frag16(etswz + ((size_t)(32 + ks) * 32 + lane) * 16);
      c0 = __builtin_amdgcn_wmma_f32_16x16x32_bf16(false, a, false, b0, (short)0, c0, false, false);
      c1 = __builtin_amdgcn_wmma_f32_16x16x32_bf16(false, a, false, b1, (short)0, c1, false, false);
    }
    #pragma unroll
    for (int j = 0; j < 8; ++j) {
      int orow = tile * 16 + j + hi * 8;
      int ob = orow / M;
      int om = orow - ob * M;
      float add = smaxRow[j + hi * 8] + tmax;
      size_t eidx = ((size_t)ob * N_DIM + parentStart + om) * L_DIM;
      size_t didx = ((size_t)ob * M + om) * L_DIM;
      dst[didx + m16]      = emission[eidx + m16]      + __logf(c0[j]) + add;
      dst[didx + 16 + m16] = emission[eidx + 16 + m16] + __logf(c1[j]) + add;
    }
  }
}

__global__ __launch_bounds__(128) void level_kernel(
    const float* __restrict__ src, int srcB, int srcOff,
    const float* __restrict__ emission,
    const unsigned short* __restrict__ etswz,
    const float* __restrict__ tmaxp,
    float* __restrict__ dst, int M, int parentStart)
{
  __shared__ float smax[4][16];
  int wave = threadIdx.x >> 5, lane = threadIdx.x & 31;
  int tile = blockIdx.x * 4 + wave;
  bool active = tile < M;                    // #tiles == M because B==16
  float tmax = tmaxp[0];
  level_body(active, tile, lane, src, srcB, srcOff, emission, etswz, tmax,
             dst, M, parentStart, &smax[wave][0]);
}

// Fused tail: levels M=32,16,8,4,2,1 in one 32-wave block (saves 5 launches).
__global__ __launch_bounds__(1024) void tail_kernel(
    const float* __restrict__ src0, int srcB0,
    const float* __restrict__ emission,
    const unsigned short* __restrict__ etswz,
    const float* __restrict__ tmaxp,
    float* __restrict__ fb0, float* __restrict__ fb1, float* __restrict__ out)
{
  __shared__ float smax[32][16];
  int wave = threadIdx.x >> 5, lane = threadIdx.x & 31;
  float tmax = tmaxp[0];
  const float* src = src0;
  int srcB = srcB0, srcOff = 0, ls = 63;
  for (int lev = 0; lev < 6; ++lev) {
    int M = 32 >> lev;
    int parent = (ls - 1) >> 1;
    float* dst = (M == 1) ? out : ((lev & 1) ? fb1 : fb0);
    bool active = wave < M;
    level_body(active, wave, lane, src, srcB, srcOff, emission, etswz, tmax,
               dst, M, parent, &smax[wave][0]);
    __syncthreads();
    src = dst; srcB = M * L_DIM; srcOff = 0;
    ls = parent;
  }
}

// ---------------------------------------------------------------------------
extern "C" void kernel_launch(void* const* d_in, const int* in_sizes, int n_in,
                              void* d_out, int out_size, void* d_ws, size_t ws_size,
                              hipStream_t stream) {
  (void)in_sizes; (void)n_in; (void)out_size; (void)ws_size;
  const float* h     = (const float*)d_in[0];
  const float* W     = (const float*)d_in[1];
  const float* bpred = (const float*)d_in[2];
  const float* trans = (const float*)d_in[3];

  char* ws = (char*)d_ws;
  size_t off = 0;
  auto take = [&](size_t bytes) -> void* {
    void* p = ws + off;
    off += (bytes + 255) & ~(size_t)255;
    return p;
  };
  float* emission = (float*)take((size_t)ROWS_TOTAL * L_DIM * sizeof(float)); // 16.8 MB
  float* bufA = (float*)take((size_t)B_DIM * 2048 * L_DIM * sizeof(float));   // 4 MB
  float* bufB = (float*)take((size_t)B_DIM * 2048 * L_DIM * sizeof(float));   // 4 MB
  float* fb0  = (float*)take((size_t)B_DIM * 32 * L_DIM * sizeof(float));
  float* fb1  = (float*)take((size_t)B_DIM * 32 * L_DIM * sizeof(float));
  unsigned short* wswz  = (unsigned short*)take(16384 * sizeof(unsigned short));
  unsigned short* etswz = (unsigned short*)take(32768 * sizeof(unsigned short));
  float* tmaxp = (float*)take(256);

  wswz_kernel<<<dim3(8), dim3(128), 0, stream>>>(W, wswz);
  et_kernel<<<dim3(1), dim3(1024), 0, stream>>>(trans, etswz, tmaxp);
  emission_kernel<<<dim3((ROWS_TOTAL / 16 + 3) / 4), dim3(128), 0, stream>>>(h, wswz, bpred, emission);

  const float* src = emission;
  int srcB = N_DIM * L_DIM;
  int srcOff = NLEAF - 1;
  int level_start = NLEAF - 1;
  int pp = 0;
  for (int M = 2048; M >= 64; M >>= 1) {
    int parent = (level_start - 1) >> 1;
    float* dst = pp ? bufB : bufA;
    int grid = (M + 3) / 4;                  // one wave per 16-node tile; tiles == M
    level_kernel<<<dim3(grid), dim3(128), 0, stream>>>(src, srcB, srcOff, emission,
                                                       etswz, tmaxp, dst, M, parent);
    src = dst; srcB = M * L_DIM; srcOff = 0;
    level_start = parent; pp ^= 1;
  }
  // level_start == 63 here; tail fuses M = 32..1 and writes d_out (B x L).
  tail_kernel<<<dim3(1), dim3(1024), 0, stream>>>(src, 64 * L_DIM, emission, etswz,
                                                  tmaxp, fb0, fb1, (float*)d_out);
}
